// Net_12309376270854
// MI455X (gfx1250) — matrix operand order repaired and checked by reading, not compiled, not run on hardware
//
#include <hip/hip_runtime.h>
#include <hip/hip_bf16.h>
#include <math.h>

// ---------------------------------------------------------------------------
// CDNA5 (gfx1250) wave32 WMMA bf16 helpers
// ---------------------------------------------------------------------------
typedef __attribute__((ext_vector_type(16))) __bf16          v16bf;
typedef __attribute__((ext_vector_type(16))) unsigned short  v16us;
typedef __attribute__((ext_vector_type(8)))  float           v8f;
typedef __attribute__((ext_vector_type(4)))  int             v4i;

union BF16Frag { v16us us; v16bf bf; };

__device__ __forceinline__ unsigned short f2bf(float f) {
  unsigned u = __float_as_uint(f);
  unsigned r = (u + 0x7FFFu + ((u >> 16) & 1u)) >> 16; // RNE
  return (unsigned short)r;
}
__device__ __forceinline__ float bf2f(unsigned short h) {
  return __uint_as_float(((unsigned)h) << 16);
}

// A-matrix (16x32 bf16) fragment from a row-major ushort buffer (LDS/global).
// Lanes 0-15 row=lane, K = e; lanes 16-31: +16 (contiguous, matches load_bT).
__device__ __forceinline__ v16bf load_a_rm(const unsigned short* a, int ldk,
                                           int kbase, int lane) {
  BF16Frag f;
  const unsigned short* row =
      a + (size_t)(lane & 15) * ldk + kbase + ((lane & 16) ? 16 : 0);
#pragma unroll
  for (int e = 0; e < 16; ++e) f.us[e] = row[e];
  return f.bf;
}

// B-matrix (32x16 bf16) fragment from TRANSPOSED weights Wt[Nout][Din].
// Lane = column n; element e holds K = kbase + e (+16 for lanes 16-31).
// Contiguous 32B per lane -> two global_load_b128.
__device__ __forceinline__ v16bf load_bT(const unsigned short* wT, int din,
                                         int kbase, int n0, int lane) {
  BF16Frag f;
  const unsigned short* p =
      wT + (size_t)(n0 + (lane & 15)) * din + kbase + ((lane & 16) ? 16 : 0);
#pragma unroll
  for (int e = 0; e < 16; ++e) f.us[e] = p[e];
  return f.bf;
}

// Guarded variant (classifier edge tile: Nout=1000).
__device__ __forceinline__ v16bf load_bT_g(const unsigned short* wT, int din,
                                           int kbase, int n0, int lane, int nmax) {
  BF16Frag f;
  int n = n0 + (lane & 15);
  if (n < nmax) {
    const unsigned short* p =
        wT + (size_t)n * din + kbase + ((lane & 16) ? 16 : 0);
#pragma unroll
    for (int e = 0; e < 16; ++e) f.us[e] = p[e];
  } else {
#pragma unroll
    for (int e = 0; e < 16; ++e) f.us[e] = 0;
  }
  return f.bf;
}

__device__ __forceinline__ v8f wmma_bf16(v16bf a, v16bf b, v8f c) {
  return __builtin_amdgcn_wmma_f32_16x16x32_bf16(false, a, false, b,
                                                 (short)0, c, false, false);
}

// ---------------------------------------------------------------------------
// Problem constants
// ---------------------------------------------------------------------------
#define BB   128
#define NP   196      // 14*14 patches
#define DM   384
#define KK   16
#define EE   8
#define PF   768      // 3*16*16 patch features
#define NCLS 1000
#define FEAT 6144     // KK*DM

// ---------------------------------------------------------------------------
// 0) f32 -> bf16 transposed conversion: src (rows x cols) -> dst (cols x rows)
// ---------------------------------------------------------------------------
__global__ void f2bf_t_kernel(const float* __restrict__ src,
                              unsigned short* __restrict__ dst, int rows, int cols) {
  int i = blockIdx.x * blockDim.x + threadIdx.x;
  if (i < rows * cols) {
    int k = i / cols, n = i % cols;
    dst[(size_t)n * rows + k] = f2bf(src[i]);
  }
}

// ---------------------------------------------------------------------------
// 1) Patch extraction + projection (WMMA) + LayerNorm, fused.
//    grid = (13 Mtiles, 128 batch), block = 256 (8 waves x 3 N-tiles).
// ---------------------------------------------------------------------------
__global__ void __launch_bounds__(256)
patch_ln_kernel(const float* __restrict__ x, const unsigned short* __restrict__ pwT,
                const float* __restrict__ pb, const float* __restrict__ ln_g,
                const float* __restrict__ ln_b, unsigned short* __restrict__ pn_bf) {
  __shared__ unsigned short a_s[16 * PF];   // 24 KB bf16 activation tile
  __shared__ float          p_s[16 * DM];   // 24 KB f32 projected tile
  int mt = blockIdx.x, b = blockIdx.y;
  // Build 16x768 patch tile (row = patch, col = c*256 + ph*16 + pw)
  for (int i = threadIdx.x; i < 16 * PF; i += 256) {
    int r = i / PF, f = i % PF;
    int n = mt * 16 + r;
    float v = 0.f;
    if (n < NP) {
      int hp = n / 14, wp = n % 14;
      int c = f >> 8, rem = f & 255, ph = rem >> 4, pwi = rem & 15;
      v = x[(((size_t)b * 3 + c) * 224 + hp * 16 + ph) * 224 + wp * 16 + pwi];
    }
    a_s[i] = f2bf(v);
  }
  __syncthreads();
  int lane = threadIdx.x & 31, wave = threadIdx.x >> 5;
  for (int t = wave * 3; t < wave * 3 + 3; ++t) {      // 24 N-tiles / 8 waves
    v8f acc = {};
    for (int kc = 0; kc < PF / 32; ++kc) {
      v16bf a  = load_a_rm(a_s, PF, kc * 32, lane);
      v16bf bb = load_bT(pwT, PF, kc * 32, t * 16, lane);
      acc = wmma_bf16(a, bb, acc);
    }
    int col = t * 16 + (lane & 15);
    float bs = pb[col];
#pragma unroll
    for (int e = 0; e < 8; ++e) {
      int m = e + ((lane & 16) ? 8 : 0);
      p_s[m * DM + col] = acc[e] + bs;
    }
  }
  __syncthreads();
  if (threadIdx.x < 16) {                               // per-row LayerNorm
    int r = threadIdx.x, n = mt * 16 + r;
    if (n < NP) {
      float mu = 0.f;
      for (int d = 0; d < DM; ++d) mu += p_s[r * DM + d];
      mu *= (1.f / DM);
      float var = 0.f;
      for (int d = 0; d < DM; ++d) { float t = p_s[r * DM + d] - mu; var += t * t; }
      var *= (1.f / DM);
      float rinv = rsqrtf(var + 1e-5f);
      unsigned short* dst = pn_bf + ((size_t)b * NP + n) * DM;
      for (int d = 0; d < DM; ++d) {
        float v = (p_s[r * DM + d] - mu) * rinv * ln_g[d] + ln_b[d];
        dst[d] = f2bf(v);
      }
    }
  }
}

// ---------------------------------------------------------------------------
// 2) K-means (10 iters) + pos add -> clusters bf16. One block per batch item.
// ---------------------------------------------------------------------------
__global__ void __launch_bounds__(256)
kmeans_kernel(const unsigned short* __restrict__ pn_bf, const float* __restrict__ pos,
              unsigned short* __restrict__ cl_bf) {
  __shared__ float cen[KK * DM];
  __shared__ float nsum[KK * DM];
  __shared__ float cnt[KK];
  int b = blockIdx.x;
  const unsigned short* pn = pn_bf + (size_t)b * NP * DM;
  for (int i = threadIdx.x; i < KK * DM; i += 256) {
    int k = i / DM, d = i % DM;
    cen[i] = bf2f(pn[(size_t)(13 * k) * DM + d]);   // linspace(0,195,16) -> 13*k
  }
  __syncthreads();
  for (int it = 0; it < 10; ++it) {
    for (int i = threadIdx.x; i < KK * DM; i += 256) nsum[i] = 0.f;
    if (threadIdx.x < KK) cnt[threadIdx.x] = 0.f;
    __syncthreads();
    for (int n = threadIdx.x; n < NP; n += 256) {
      float acc[KK];
#pragma unroll
      for (int k = 0; k < KK; ++k) acc[k] = 0.f;
      for (int d = 0; d < DM; ++d) {
        float p = bf2f(pn[(size_t)n * DM + d]);
#pragma unroll
        for (int k = 0; k < KK; ++k) { float df = p - cen[k * DM + d]; acc[k] += df * df; }
      }
      int best = 0; float bv = acc[0];
#pragma unroll
      for (int k = 1; k < KK; ++k) if (acc[k] < bv) { bv = acc[k]; best = k; }
      atomicAdd(&cnt[best], 1.f);
      for (int d = 0; d < DM; ++d)
        atomicAdd(&nsum[best * DM + d], bf2f(pn[(size_t)n * DM + d]));
    }
    __syncthreads();
    for (int i = threadIdx.x; i < KK * DM; i += 256) {
      int k = i / DM;
      if (cnt[k] > 0.f) cen[i] = nsum[i] / cnt[k];
    }
    __syncthreads();
  }
  for (int i = threadIdx.x; i < KK * DM; i += 256)
    cl_bf[(size_t)b * KK * DM + i] = f2bf(cen[i] + pos[i]);
}

// ---------------------------------------------------------------------------
// 3) Generic 16x384 slab matmul (WMMA): out = act(A_bf @ W^T_bf + bias).
//    One block per batch item; A slab staged via async global->LDS DMA.
// ---------------------------------------------------------------------------
__global__ void __launch_bounds__(256)
slab_matmul_kernel(const unsigned short* __restrict__ A_bf,
                   const unsigned short* __restrict__ WT_bf,
                   const float* __restrict__ bias,
                   float* __restrict__ out_f32, unsigned short* __restrict__ out_bf,
                   int act) {
  __shared__ unsigned short a_s[16 * DM];
  int b = blockIdx.x;
  const unsigned short* A = A_bf + (size_t)b * KK * DM;
#if __has_builtin(__builtin_amdgcn_global_load_async_to_lds_b128)
  // CDNA5 async DMA: 16*384 bf16 = 768 x 16B transfers, ASYNCcnt-tracked.
  for (int i = threadIdx.x; i < (KK * DM) / 8; i += 256) {
    __builtin_amdgcn_global_load_async_to_lds_b128(
        (__attribute__((address_space(1))) v4i*)(A + (size_t)i * 8),
        (__attribute__((address_space(3))) v4i*)(&a_s[i * 8]), 0, 0);
  }
#if __has_builtin(__builtin_amdgcn_s_wait_asynccnt)
  __builtin_amdgcn_s_wait_asynccnt(0);
#else
  asm volatile("s_wait_asynccnt 0" ::: "memory");
#endif
#else
  for (int i = threadIdx.x; i < KK * DM; i += 256) a_s[i] = A[i];
#endif
  __syncthreads();
  int lane = threadIdx.x & 31, wave = threadIdx.x >> 5;
  for (int t = wave * 3; t < wave * 3 + 3; ++t) {
    v8f acc = {};
    for (int kc = 0; kc < DM / 32; ++kc) {
      v16bf a  = load_a_rm(a_s, DM, kc * 32, lane);
      v16bf bb = load_bT(WT_bf, DM, kc * 32, t * 16, lane);
      acc = wmma_bf16(a, bb, acc);
    }
    int col = t * 16 + (lane & 15);
    float bs = bias[col];
#pragma unroll
    for (int e = 0; e < 8; ++e) {
      int m = e + ((lane & 16) ? 8 : 0);
      float v = acc[e] + bs;
      if (act == 1) v = 0.5f * v * (1.f + erff(v * 0.70710678118654752f)); // exact GELU
      size_t idx = (size_t)b * KK * DM + (size_t)m * DM + col;
      if (out_f32) out_f32[idx] = v;
      if (out_bf)  out_bf[idx]  = f2bf(v);
    }
  }
}

// ---------------------------------------------------------------------------
// 4) Attention (16x16 softmax) + hypergraph Laplacian. One block per batch.
//    softmax(hw, axis=1) => Dd == 1, so M == Hm and L = I - Hm Hm^T.
// ---------------------------------------------------------------------------
__global__ void __launch_bounds__(256)
attn_hg_kernel(const float* __restrict__ Qf, const float* __restrict__ Ktf,
               const float* __restrict__ Vf, const float* __restrict__ ew,
               const float* __restrict__ eb,
               float* __restrict__ c3_f32, unsigned short* __restrict__ c3_bf) {
  __shared__ float bufA[KK * DM];
  __shared__ float bufB[KK * DM];
  __shared__ float attn[KK * KK];
  __shared__ float hw_s[KK * EE];
  __shared__ float L_s[EE * EE];
  int b = blockIdx.x;
  size_t base = (size_t)b * KK * DM;
  for (int i = threadIdx.x; i < KK * DM; i += 256) { bufA[i] = Qf[base + i]; bufB[i] = Ktf[base + i]; }
  __syncthreads();
  { // scores
    int q = threadIdx.x >> 4, k = threadIdx.x & 15;
    float a = 0.f;
    for (int d = 0; d < DM; ++d) a += bufA[q * DM + d] * bufB[k * DM + d];
    attn[q * KK + k] = a * 0.05103103630798287f; // 1/sqrt(384)
  }
  __syncthreads();
  if (threadIdx.x < KK) { // row softmax
    int q = threadIdx.x;
    float mx = -1e30f;
    for (int k = 0; k < KK; ++k) mx = fmaxf(mx, attn[q * KK + k]);
    float s = 0.f;
    for (int k = 0; k < KK; ++k) { float e = expf(attn[q * KK + k] - mx); attn[q * KK + k] = e; s += e; }
    float inv = 1.f / s;
    for (int k = 0; k < KK; ++k) attn[q * KK + k] *= inv;
  }
  __syncthreads();
  for (int i = threadIdx.x; i < KK * DM; i += 256) bufA[i] = Vf[base + i]; // V
  __syncthreads();
  for (int i = threadIdx.x; i < KK * DM; i += 256) { // clusters = attn @ V
    int m = i / DM, d = i % DM;
    float a = 0.f;
#pragma unroll
    for (int k = 0; k < KK; ++k) a += attn[m * KK + k] * bufA[k * DM + d];
    bufB[i] = a;
  }
  __syncthreads();
  if (threadIdx.x < KK * EE) { // hyperedge logits
    int k = threadIdx.x >> 3, e = threadIdx.x & 7;
    float a = eb[e];
    for (int d = 0; d < DM; ++d) a += bufB[k * DM + d] * ew[d * EE + e];
    hw_s[k * EE + e] = a;
  }
  __syncthreads();
  if (threadIdx.x < EE) { // softmax over clusters (axis=1)
    int e = threadIdx.x;
    float mx = -1e30f;
    for (int k = 0; k < KK; ++k) mx = fmaxf(mx, hw_s[k * EE + e]);
    float s = 0.f;
    for (int k = 0; k < KK; ++k) { float v = expf(hw_s[k * EE + e] - mx); hw_s[k * EE + e] = v; s += v; }
    float inv = 1.f / s;
    for (int k = 0; k < KK; ++k) hw_s[k * EE + e] *= inv;
  }
  __syncthreads();
  if (threadIdx.x < EE * EE) { // L = I - Hm Hm^T
    int e = threadIdx.x >> 3, f = threadIdx.x & 7;
    float s = 0.f;
#pragma unroll
    for (int k = 0; k < KK; ++k) s += hw_s[k * EE + e] * hw_s[k * EE + f];
    L_s[e * EE + f] = ((e == f) ? 1.f : 0.f) - s;
  }
  __syncthreads();
  for (int i = threadIdx.x; i < KK * DM; i += 256) { // padded L @ clusters
    int m = i / DM, d = i % DM;
    float v = 0.f;
    if (m < EE) {
#pragma unroll
      for (int j = 0; j < EE; ++j) v += L_s[m * EE + j] * bufB[j * DM + d];
    }
    c3_f32[base + i] = v;
    c3_bf[base + i] = f2bf(v);
  }
}

// ---------------------------------------------------------------------------
// 5) Cosine-similarity mask (cross-batch OR via plain stores of 1).
// ---------------------------------------------------------------------------
__global__ void __launch_bounds__(256)
sim_mask_kernel(const float* __restrict__ c6, int* __restrict__ mask) {
  __shared__ float nc[KK * DM];
  __shared__ float rnorm[KK];
  int b = blockIdx.x;
  size_t base = (size_t)b * KK * DM;
  for (int i = threadIdx.x; i < KK * DM; i += 256) nc[i] = c6[base + i];
  __syncthreads();
  if (threadIdx.x < KK) {
    int r = threadIdx.x;
    float s = 0.f;
    for (int d = 0; d < DM; ++d) { float v = nc[r * DM + d]; s += v * v; }
    rnorm[r] = 1.f / fmaxf(sqrtf(s), 1e-12f);
  }
  __syncthreads();
  if (threadIdx.x < 120) { // 16 choose 2 pairs
    int t = threadIdx.x, i = 0, rem = t;
    while (rem >= 15 - i) { rem -= 15 - i; ++i; }
    int j = i + 1 + rem;
    float s = 0.f;
    for (int d = 0; d < DM; ++d) s += nc[i * DM + d] * nc[j * DM + d];
    s *= rnorm[i] * rnorm[j];
    if (s > 0.9f) mask[i * KK + j] = 1;
  }
}

// ---------------------------------------------------------------------------
// 6) Sequential merge + global feature + gated feedback -> bf16 features.
// ---------------------------------------------------------------------------
__global__ void __launch_bounds__(256)
merge_head_kernel(const float* __restrict__ c6, const int* __restrict__ mask,
                  const float* __restrict__ fbw, const float* __restrict__ fbb,
                  const float* __restrict__ fgw, const float* __restrict__ fgb,
                  unsigned short* __restrict__ feat_bf) {
  __shared__ float c[KK * DM];
  __shared__ int   m_s[KK * KK];
  __shared__ float g_s[DM];
  __shared__ float red[256];
  __shared__ float gate_s;
  int b = blockIdx.x;
  size_t base = (size_t)b * KK * DM;
  for (int i = threadIdx.x; i < KK * DM; i += 256) c[i] = c6[base + i];
  if (threadIdx.x < KK * KK) m_s[threadIdx.x] = mask[threadIdx.x];
  __syncthreads();
  for (int i = 0; i < KK; ++i)
    for (int j = i + 1; j < KK; ++j) {
      if (m_s[i * KK + j]) {              // uniform across block
        for (int d = threadIdx.x; d < DM; d += 256) {
          float mg = 0.5f * (c[i * DM + d] + c[j * DM + d]);
          c[i * DM + d] = mg; c[j * DM + d] = mg;
        }
        __syncthreads();
      }
    }
  __syncthreads();
  for (int d = threadIdx.x; d < DM; d += 256) { // g = mean over clusters
    float s = 0.f;
#pragma unroll
    for (int k = 0; k < KK; ++k) s += c[k * DM + d];
    g_s[d] = s * (1.f / KK);
  }
  __syncthreads();
  float part = 0.f; // gate = sigmoid(g . fgw + fgb)
  for (int d = threadIdx.x; d < DM; d += 256) part += g_s[d] * fgw[d];
  red[threadIdx.x] = part;
  __syncthreads();
  for (int s = 128; s > 0; s >>= 1) {
    if (threadIdx.x < s) red[threadIdx.x] += red[threadIdx.x + s];
    __syncthreads();
  }
  if (threadIdx.x == 0) gate_s = 1.f / (1.f + expf(-(red[0] + fgb[0])));
  __syncthreads();
  for (int d2 = threadIdx.x; d2 < DM; d2 += 256) {
    float a = fbb[d2];
    for (int d = 0; d < DM; ++d) a += g_s[d] * fbw[d * DM + d2];
    float add = a * gate_s;
    for (int k = 0; k < KK; ++k)
      feat_bf[(size_t)b * FEAT + k * DM + d2] = f2bf(c[k * DM + d2] + add);
  }
}

// ---------------------------------------------------------------------------
// 7) Classifier (WMMA): (128 x 6144) @ (6144 x 1000) + cb -> d_out.
//    grid = (8 Mtiles, 8), 8 waves/block -> 64 N-tile slots for 63 tiles.
// ---------------------------------------------------------------------------
__global__ void __launch_bounds__(256)
classifier_kernel(const unsigned short* __restrict__ feat_bf,
                  const unsigned short* __restrict__ cwT_bf,
                  const float* __restrict__ cb, float* __restrict__ out) {
  int lane = threadIdx.x & 31, wave = threadIdx.x >> 5;
  int nt = blockIdx.y * 8 + wave;
  if (nt >= (NCLS + 15) / 16) return;            // uniform per wave
  int m0 = blockIdx.x * 16, n0 = nt * 16;
  int col = n0 + (lane & 15);
  v8f acc = {};
  for (int kc = 0; kc < FEAT / 32; ++kc) {
    v16bf a  = load_a_rm(feat_bf + (size_t)m0 * FEAT, FEAT, kc * 32, lane);
    v16bf bb = load_bT_g(cwT_bf, FEAT, kc * 32, n0, lane, NCLS);
    acc = wmma_bf16(a, bb, acc);
  }
  if (col < NCLS) {
    float bs = cb[col];
#pragma unroll
    for (int e = 0; e < 8; ++e) {
      int m = e + ((lane & 16) ? 8 : 0);
      out[(size_t)(m0 + m) * NCLS + col] = acc[e] + bs;
    }
  }
}

// ---------------------------------------------------------------------------
// Host-side orchestration
// ---------------------------------------------------------------------------
extern "C" void kernel_launch(void* const* d_in, const int* in_sizes, int n_in,
                              void* d_out, int out_size, void* d_ws, size_t ws_size,
                              hipStream_t stream) {
  const float* x   = (const float*)d_in[0];
  const float* pw  = (const float*)d_in[1];
  const float* pb  = (const float*)d_in[2];
  const float* lng = (const float*)d_in[3];
  const float* lnb = (const float*)d_in[4];
  const float* pos = (const float*)d_in[5];
  const float* qw  = (const float*)d_in[6];
  const float* qb  = (const float*)d_in[7];
  const float* kw  = (const float*)d_in[8];
  const float* kb  = (const float*)d_in[9];
  const float* vw  = (const float*)d_in[10];
  const float* vb  = (const float*)d_in[11];
  const float* ew  = (const float*)d_in[12];
  const float* eb  = (const float*)d_in[13];
  const float* nw  = (const float*)d_in[14];
  const float* nb  = (const float*)d_in[15];
  const float* m1w = (const float*)d_in[16];
  const float* m1b = (const float*)d_in[17];
  const float* m2w = (const float*)d_in[18];
  const float* m2b = (const float*)d_in[19];
  const float* fbw = (const float*)d_in[20];
  const float* fbb = (const float*)d_in[21];
  const float* fgw = (const float*)d_in[22];
  const float* fgb = (const float*)d_in[23];
  const float* cw  = (const float*)d_in[24];
  const float* cb  = (const float*)d_in[25];
  float* out = (float*)d_out;

  char* ws = (char*)d_ws;
  size_t off = 0;
  auto take = [&](size_t bytes) -> void* {
    void* p = ws + off;
    off = (off + bytes + 255) & ~(size_t)255;
    return p;
  };
  // All weights stored TRANSPOSED bf16: Wt[Nout][Din]
  unsigned short* pwT  = (unsigned short*)take((size_t)PF * DM * 2);
  unsigned short* qwT  = (unsigned short*)take((size_t)DM * DM * 2);
  unsigned short* kwT  = (unsigned short*)take((size_t)DM * DM * 2);
  unsigned short* vwT  = (unsigned short*)take((size_t)DM * DM * 2);
  unsigned short* nwT  = (unsigned short*)take((size_t)DM * DM * 2);
  unsigned short* m1wT = (unsigned short*)take((size_t)DM * DM * 2);
  unsigned short* m2wT = (unsigned short*)take((size_t)DM * DM * 2);
  unsigned short* cwT  = (unsigned short*)take((size_t)FEAT * NCLS * 2);
  unsigned short* pn_bf  = (unsigned short*)take((size_t)BB * NP * DM * 2);
  unsigned short* clA_bf = (unsigned short*)take((size_t)BB * KK * DM * 2);
  unsigned short* clB_bf = (unsigned short*)take((size_t)BB * KK * DM * 2);
  unsigned short* ft_bf  = (unsigned short*)take((size_t)BB * FEAT * 2);
  float* Qf    = (float*)take((size_t)BB * KK * DM * 4);
  float* Ktf   = (float*)take((size_t)BB * KK * DM * 4);
  float* Vf    = (float*)take((size_t)BB * KK * DM * 4);
  float* bufC0 = (float*)take((size_t)BB * KK * DM * 4);
  float* bufC1 = (float*)take((size_t)BB * KK * DM * 4);
  int*   mask  = (int*)take(KK * KK * sizeof(int));
  (void)in_sizes; (void)n_in; (void)out_size; (void)ws_size;

  // 0) transposed weight conversions + mask clear
  f2bf_t_kernel<<<(PF * DM + 255) / 256, 256, 0, stream>>>(pw, pwT, PF, DM);
  f2bf_t_kernel<<<(DM * DM + 255) / 256, 256, 0, stream>>>(qw, qwT, DM, DM);
  f2bf_t_kernel<<<(DM * DM + 255) / 256, 256, 0, stream>>>(kw, kwT, DM, DM);
  f2bf_t_kernel<<<(DM * DM + 255) / 256, 256, 0, stream>>>(vw, vwT, DM, DM);
  f2bf_t_kernel<<<(DM * DM + 255) / 256, 256, 0, stream>>>(nw, nwT, DM, DM);
  f2bf_t_kernel<<<(DM * DM + 255) / 256, 256, 0, stream>>>(m1w, m1wT, DM, DM);
  f2bf_t_kernel<<<(DM * DM + 255) / 256, 256, 0, stream>>>(m2w, m2wT, DM, DM);
  f2bf_t_kernel<<<(FEAT * NCLS + 255) / 256, 256, 0, stream>>>(cw, cwT, FEAT, NCLS);
  (void)hipMemsetAsync(mask, 0, KK * KK * sizeof(int), stream);

  // 1) patch embed + LN (WMMA)
  patch_ln_kernel<<<dim3(13, BB), 256, 0, stream>>>(x, pwT, pb, lng, lnb, pn_bf);
  // 2) k-means + pos
  kmeans_kernel<<<BB, 256, 0, stream>>>(pn_bf, pos, clA_bf);
  // 3) Q, K, V (WMMA)
  slab_matmul_kernel<<<BB, 256, 0, stream>>>(clA_bf, qwT, qb, Qf, nullptr, 0);
  slab_matmul_kernel<<<BB, 256, 0, stream>>>(clA_bf, kwT, kb, Ktf, nullptr, 0);
  slab_matmul_kernel<<<BB, 256, 0, stream>>>(clA_bf, vwT, vb, Vf, nullptr, 0);
  // 4) attention + hypergraph Laplacian
  attn_hg_kernel<<<BB, 256, 0, stream>>>(Qf, Ktf, Vf, ew, eb, bufC0, clB_bf);
  // 5) nw, MLP (WMMA)
  slab_matmul_kernel<<<BB, 256, 0, stream>>>(clB_bf, nwT, nb, bufC1, clA_bf, 0);
  slab_matmul_kernel<<<BB, 256, 0, stream>>>(clA_bf, m1wT, m1b, nullptr, clB_bf, 1);
  slab_matmul_kernel<<<BB, 256, 0, stream>>>(clB_bf, m2wT, m2b, bufC1, nullptr, 0);
  // 6) cross-batch similarity mask, then merges + head
  sim_mask_kernel<<<BB, 256, 0, stream>>>(bufC1, mask);
  merge_head_kernel<<<BB, 256, 0, stream>>>(bufC1, mask, fbw, fbb, fgw, fgb, ft_bf);
  // 7) classifier (WMMA)
  classifier_kernel<<<dim3(8, 8), 256, 0, stream>>>(ft_bf, cwT, cb, out);
}
